// TransformerNet_35167192220488
// MI455X (gfx1250) — matrix operand order repaired and checked by reading, not compiled, not run on hardware
//
#include <hip/hip_runtime.h>

#define N_NODES 50000
#define N_EDGES 1600000
#define IN_FEAT 128
#define NHEAD   4
#define HDIM    16
#define FEAT    64          // NHEAD * HDIM
#define MTILES  3125        // N_NODES / 16

typedef __attribute__((ext_vector_type(16))) _Float16 v16h;
typedef __attribute__((ext_vector_type(8)))  _Float16 v8h;
typedef __attribute__((ext_vector_type(8)))  float    v8f;

// ---------------- helpers ----------------

__device__ __forceinline__ void atomicMaxF(float* addr, float val) {
  // monotone-bit-pattern trick; addr must be initialized to -inf
  if (val >= 0.0f) atomicMax((int*)addr, __float_as_int(val));
  else             atomicMin((unsigned int*)addr, __float_as_uint(val));
}

__device__ __forceinline__ void atomicAddF(float* addr, float val) {
  // relaxed, agent-scope fadd; lowers to non-returning global_atomic_add_f32
  (void)__hip_atomic_fetch_add(addr, val, __ATOMIC_RELAXED, __HIP_MEMORY_SCOPE_AGENT);
}

// ---------------- conversion kernels ----------------

__global__ void cvt_f32_f16(const float* __restrict__ in, _Float16* __restrict__ out, int n) {
  int i = blockIdx.x * blockDim.x + threadIdx.x;
  if (i < n) out[i] = (_Float16)in[i];
}

// W: [K x 64] row-major f32  ->  Wt: [64 x K] row-major f16 (transposed)
__global__ void cvt_w_t(const float* __restrict__ W, _Float16* __restrict__ Wt, int K) {
  int t = blockIdx.x * blockDim.x + threadIdx.x;
  if (t < K * FEAT) {
    int kk = t >> 6;        // row in W
    int nn = t & 63;        // col in W
    Wt[(size_t)nn * K + kk] = (_Float16)W[t];
  }
}

// ---------------- WMMA projection GEMM ----------------
// out[p] = A(f16)[N x K] @ W[p](f16)[K x 64] + b[p], p in {q,k,v,skip}
// One wave per 16x16 output tile. blockIdx.y = proj*4 + ntile.

struct GemmArgs {
  const _Float16* A;
  const _Float16* W0; const _Float16* W1; const _Float16* W2; const _Float16* W3;
  const float* b0; const float* b1; const float* b2; const float* b3;
  float* o0; float* o1; float* o2; float* o3;
};

template <int K>
__global__ __launch_bounds__(256) void gemm_qkvs(GemmArgs ga) {
  const int wave  = threadIdx.x >> 5;
  const int lane  = threadIdx.x & 31;
  const int mtile = blockIdx.x * 8 + wave;
  if (mtile >= MTILES) return;                 // wave-uniform: EXEC stays all-1s
  const int proj  = blockIdx.y >> 2;
  const int ntile = blockIdx.y & 3;

  const _Float16* Wt  = (proj == 0) ? ga.W0 : (proj == 1) ? ga.W1 : (proj == 2) ? ga.W2 : ga.W3;
  const float*    bia = (proj == 0) ? ga.b0 : (proj == 1) ? ga.b1 : (proj == 2) ? ga.b2 : ga.b3;
  float*          out = (proj == 0) ? ga.o0 : (proj == 1) ? ga.o1 : (proj == 2) ? ga.o2 : ga.o3;

  const int hi  = lane >> 4;     // half-wave select (ISA A/B K-split)
  const int l15 = lane & 15;
  const int row = mtile * 16 + l15;      // A row this lane owns
  const int col = ntile * 16 + l15;      // output column this lane owns

  const _Float16* arow = ga.A + (size_t)row * K;
  const _Float16* wrow = Wt   + (size_t)col * K + hi * 16;   // contiguous 16 halves per step

  v8f c = {};
#pragma unroll
  for (int k0 = 0; k0 < K; k0 += 32) {
    // A fragment (contiguous K order): halves 0..7 -> K = k0+hi*16+i,
    //                                  halves 8..15 -> K = k0+hi*16+8+i
    v8h alo = *(const v8h*)(arow + k0 + hi * 16);
    v8h ahi = *(const v8h*)(arow + k0 + hi * 16 + 8);
    v16h a;
#pragma unroll
    for (int i = 0; i < 8; ++i) { a[i] = alo[i]; a[i + 8] = ahi[i]; }
    // B fragment: lanes 0-15 hold K=k0..k0+15, lanes 16-31 hold K=k0+16..k0+31
    v16h b = *(const v16h*)(wrow + k0);
    c = __builtin_amdgcn_wmma_f32_16x16x32_f16(false, a, false, b, (short)0, c, false, false);
  }

  const float bv = bia[col];
#pragma unroll
  for (int r = 0; r < 8; ++r) {
    int m = mtile * 16 + r + hi * 8;     // D layout: VGPR r -> row r (+8 for upper half-wave)
    out[(size_t)m * FEAT + col] = c[r] + bv;
  }
}

// ---------------- per-layer init ----------------

__global__ void init_layer(float* __restrict__ amax, float* __restrict__ denom,
                           float* __restrict__ agg, int nh, int nagg) {
  int i = blockIdx.x * blockDim.x + threadIdx.x;
  if (i < nh) { amax[i] = -__builtin_inff(); denom[i] = 0.0f; }
  if (i < nagg) agg[i] = 0.0f;
}

// ---------------- edge pass 1: logits + segment max ----------------
// one thread per edge, all 4 heads (one dst/src index load feeds 4 heads)

__global__ __launch_bounds__(256) void edge_logits(const int* __restrict__ src,
                                                   const int* __restrict__ dst,
                                                   const float* __restrict__ q,
                                                   const float* __restrict__ k,
                                                   float* __restrict__ alpha,
                                                   float* __restrict__ amax, int ne) {
  int e = blockIdx.x * blockDim.x + threadIdx.x;
  if (e >= ne) return;
  __builtin_prefetch(src + e + 4096, 0, 0);
  __builtin_prefetch(dst + e + 4096, 0, 0);
  int s = src[e], d = dst[e];
  const float4* qp = (const float4*)(q + (size_t)d * FEAT);
  const float4* kp = (const float4*)(k + (size_t)s * FEAT);
#pragma unroll
  for (int h = 0; h < NHEAD; ++h) {
    float acc = 0.0f;
#pragma unroll
    for (int i = 0; i < 4; ++i) {
      float4 a = qp[h * 4 + i], b = kp[h * 4 + i];
      acc += a.x * b.x + a.y * b.y + a.z * b.z + a.w * b.w;
    }
    acc *= 0.25f;                     // 1/sqrt(D), D = 16
    alpha[(size_t)e * NHEAD + h] = acc;
    atomicMaxF(&amax[d * NHEAD + h], acc);
  }
}

// ---------------- edge pass 2: exp + segment sum ----------------
// thread per (edge, head)

__global__ __launch_bounds__(256) void edge_expsum(const int* __restrict__ dst,
                                                   float* __restrict__ alpha,
                                                   const float* __restrict__ amax,
                                                   float* __restrict__ denom, int eh) {
  int t = blockIdx.x * blockDim.x + threadIdx.x;
  if (t >= eh) return;
  int e = t >> 2, h = t & 3;
  __builtin_prefetch(alpha + t + 4096, 0, 0);
  int d = dst[e];
  float ex = __expf(alpha[t] - amax[d * NHEAD + h]);
  alpha[t] = ex;                      // reuse buffer: now holds exp()
  atomicAddF(&denom[d * NHEAD + h], ex);
}

// ---------------- edge pass 3: weighted scatter-aggregate ----------------
// 32 threads per edge; each thread handles 2 consecutive feature slots

__global__ __launch_bounds__(256) void edge_aggr(const int* __restrict__ src,
                                                 const int* __restrict__ dst,
                                                 const float* __restrict__ alpha,
                                                 const float* __restrict__ denom,
                                                 const float* __restrict__ v,
                                                 float* __restrict__ agg, int nthreads) {
  int t = blockIdx.x * blockDim.x + threadIdx.x;
  if (t >= nthreads) return;
  int e  = t >> 5;
  int li = (t & 31) * 2;              // feature pair index, 0..62 (stays within a head)
  int h  = li >> 4;
  int s = src[e], d = dst[e];
  float w = alpha[(size_t)e * NHEAD + h] / denom[d * NHEAD + h];
  float2 vv = *(const float2*)(v + (size_t)s * FEAT + li);
  atomicAddF(&agg[(size_t)d * FEAT + li],     w * vv.x);
  atomicAddF(&agg[(size_t)d * FEAT + li + 1], w * vv.y);
}

// ---------------- skip + relu + f16 convert for next layer ----------------

__global__ void skip_relu_cvt(const float* __restrict__ agg, const float* __restrict__ skp,
                              _Float16* __restrict__ h16, int n) {
  int i = blockIdx.x * blockDim.x + threadIdx.x;
  if (i < n) {
    float r = agg[i] + skp[i];
    h16[i] = (_Float16)(r > 0.0f ? r : 0.0f);
  }
}

// ---------------- final 64 -> 1 linear ----------------

__global__ void final_linear(const _Float16* __restrict__ h16, const float* __restrict__ Wf,
                             const float* __restrict__ bf, float* __restrict__ out, int n) {
  int i = blockIdx.x * blockDim.x + threadIdx.x;
  if (i >= n) return;
  const _Float16* hr = h16 + (size_t)i * FEAT;
  float acc = bf[0];
#pragma unroll
  for (int c = 0; c < FEAT; ++c) acc += (float)hr[c] * Wf[c];
  out[i] = acc;
}

// ---------------- host orchestration ----------------

extern "C" void kernel_launch(void* const* d_in, const int* in_sizes, int n_in,
                              void* d_out, int out_size, void* d_ws, size_t ws_size,
                              hipStream_t stream) {
  (void)in_sizes; (void)n_in; (void)out_size; (void)ws_size;

  const float* x  = (const float*)d_in[0];
  const int*   ei = (const int*)d_in[1];
  const int*   src = ei;
  const int*   dst = ei + N_EDGES;
  const float* Wf = (const float*)d_in[26];
  const float* bf = (const float*)d_in[27];
  float* outp = (float*)d_out;

  // ---- workspace carve-up (all offsets multiples of 256B) ----
  char* ws = (char*)d_ws;
  size_t off = 0;
  auto carve = [&](size_t bytes) { char* p = ws + off; off += (bytes + 255) & ~(size_t)255; return p; };
  _Float16* h16   = (_Float16*)carve((size_t)N_NODES * IN_FEAT * 2);  // GEMM input (K=128 or 64)
  float*    qb    = (float*)carve((size_t)N_NODES * FEAT * 4);
  float*    kb    = (float*)carve((size_t)N_NODES * FEAT * 4);
  float*    vb    = (float*)carve((size_t)N_NODES * FEAT * 4);
  float*    skipb = (float*)carve((size_t)N_NODES * FEAT * 4);
  float*    aggb  = (float*)carve((size_t)N_NODES * FEAT * 4);
  float*    alpha = (float*)carve((size_t)N_EDGES * NHEAD * 4);
  float*    amax  = (float*)carve((size_t)N_NODES * NHEAD * 4);
  float*    denom = (float*)carve((size_t)N_NODES * NHEAD * 4);
  _Float16* wt16  = (_Float16*)carve((size_t)4 * FEAT * IN_FEAT * 2); // 4 transposed weights, K<=128

  const int TB = 256;
  const int EH = N_EDGES * NHEAD;

  // x -> f16
  cvt_f32_f16<<<(N_NODES * IN_FEAT + TB - 1) / TB, TB, 0, stream>>>(x, h16, N_NODES * IN_FEAT);

  int K = IN_FEAT;
  for (int l = 0; l < 3; ++l) {
    const float* Wq = (const float*)d_in[2 + l * 8 + 0];
    const float* bq = (const float*)d_in[2 + l * 8 + 1];
    const float* Wk = (const float*)d_in[2 + l * 8 + 2];
    const float* bk = (const float*)d_in[2 + l * 8 + 3];
    const float* Wv = (const float*)d_in[2 + l * 8 + 4];
    const float* bv = (const float*)d_in[2 + l * 8 + 5];
    const float* Ws = (const float*)d_in[2 + l * 8 + 6];
    const float* bs = (const float*)d_in[2 + l * 8 + 7];

    _Float16* wq16 = wt16 + 0 * (size_t)FEAT * IN_FEAT;
    _Float16* wk16 = wt16 + 1 * (size_t)FEAT * IN_FEAT;
    _Float16* wv16 = wt16 + 2 * (size_t)FEAT * IN_FEAT;
    _Float16* ws16 = wt16 + 3 * (size_t)FEAT * IN_FEAT;

    int wgrid = (K * FEAT + TB - 1) / TB;
    cvt_w_t<<<wgrid, TB, 0, stream>>>(Wq, wq16, K);
    cvt_w_t<<<wgrid, TB, 0, stream>>>(Wk, wk16, K);
    cvt_w_t<<<wgrid, TB, 0, stream>>>(Wv, wv16, K);
    cvt_w_t<<<wgrid, TB, 0, stream>>>(Ws, ws16, K);

    GemmArgs ga;
    ga.A = h16;
    ga.W0 = wq16; ga.W1 = wk16; ga.W2 = wv16; ga.W3 = ws16;
    ga.b0 = bq;   ga.b1 = bk;   ga.b2 = bv;   ga.b3 = bs;
    ga.o0 = qb;   ga.o1 = kb;   ga.o2 = vb;   ga.o3 = skipb;

    dim3 ggrid((MTILES + 7) / 8, 16);
    if (K == 128) gemm_qkvs<128><<<ggrid, 256, 0, stream>>>(ga);
    else          gemm_qkvs<64><<<ggrid, 256, 0, stream>>>(ga);

    init_layer<<<(N_NODES * FEAT + TB - 1) / TB, TB, 0, stream>>>(
        amax, denom, aggb, N_NODES * NHEAD, N_NODES * FEAT);

    edge_logits<<<(N_EDGES + TB - 1) / TB, TB, 0, stream>>>(src, dst, qb, kb, alpha, amax, N_EDGES);
    edge_expsum<<<(EH + TB - 1) / TB, TB, 0, stream>>>(dst, alpha, amax, denom, EH);

    int aggThreads = N_EDGES * 32;
    edge_aggr<<<(aggThreads + TB - 1) / TB, TB, 0, stream>>>(src, dst, alpha, denom, vb, aggb, aggThreads);

    skip_relu_cvt<<<(N_NODES * FEAT + TB - 1) / TB, TB, 0, stream>>>(aggb, skipb, h16, N_NODES * FEAT);
    K = FEAT;  // next layers consume 64-wide f16 features
  }

  final_linear<<<(N_NODES + TB - 1) / TB, TB, 0, stream>>>(h16, Wf, bf, outp, N_NODES);
}
